// HopfieldNet_88089779241125
// MI455X (gfx1250) — matrix operand order repaired and checked by reading, not compiled, not run on hardware
//
#include <hip/hip_runtime.h>

// ---------------------------------------------------------------------------
// Hopfield fixed-point: Q <- X^T sparsemax(X Q), 16 iters.
//   X: [N=65536, D=256] fp32 (d_in[1]),  Q: [D=256, M=64] fp32 (d_in[0])
// Strategy (MI455X / gfx1250):
//   * bf16 WMMA 16x16x32 with hi/lo split (3 products) ~ fp32 accuracy,
//     ~8/3x the FLOP rate of the f32 16x16x4 WMMA path.
//   * S stored transposed [M][N] so each sparsemax column is a contiguous
//     256 KB row that fits in one WGP's 320 KB LDS; the column is DMA'd into
//     LDS by the Tensor Data Mover (tensor_load_to_lds + s_wait_tensorcnt).
//   * GEMM2 as C' = p^T X (split-K, deterministic partial reduce); C' == Qt
//     for the next iteration.
// ---------------------------------------------------------------------------

typedef __attribute__((ext_vector_type(8)))  __bf16 v8bf;
typedef __attribute__((ext_vector_type(16))) __bf16 v16bf;
typedef __attribute__((ext_vector_type(8)))  float  v8f;
typedef __attribute__((ext_vector_type(4)))  unsigned int u32x4;
typedef __attribute__((ext_vector_type(8)))  int    i32x8;
typedef __attribute__((ext_vector_type(4)))  int    i32x4;

#define WMMA_BF16(A, B, C) \
  __builtin_amdgcn_wmma_f32_16x16x32_bf16(false, (A), false, (B), (short)0, (C), false, false)

#define CAT16(lo, hi) \
  __builtin_shufflevector((lo), (hi), 0,1,2,3,4,5,6,7,8,9,10,11,12,13,14,15)

static constexpr int N = 65536;  // stored patterns
static constexpr int D = 256;    // pattern dim
static constexpr int M = 64;     // queries
static constexpr int ITERS = 16;

// ---------------------------------------------------------------------------
// Workspace layout (bytes)
// ---------------------------------------------------------------------------
static constexpr size_t OFF_XH   = 0;                         // bf16 [N][D]  32 MB
static constexpr size_t OFF_XL   = OFF_XH  + (size_t)N*D*2;   // bf16 [N][D]  32 MB
static constexpr size_t OFF_ST   = OFF_XL  + (size_t)N*D*2;   // f32  [M][N]  16 MB
static constexpr size_t OFF_PH   = OFF_ST  + (size_t)M*N*4;   // bf16 [M][N]   8 MB
static constexpr size_t OFF_PL   = OFF_PH  + (size_t)M*N*2;   // bf16 [M][N]   8 MB
static constexpr size_t OFF_PART = OFF_PL  + (size_t)M*N*2;   // f32  [64][M][D] 4 MB
static constexpr size_t OFF_QTF  = OFF_PART+ (size_t)64*M*D*4;// f32  [M][D]
static constexpr size_t OFF_QTH  = OFF_QTF + (size_t)M*D*4;   // bf16 [M][D]
static constexpr size_t OFF_QTL  = OFF_QTH + (size_t)M*D*2;   // bf16 [M][D]

// ---------------------------------------------------------------------------
// 1) Split X into bf16 hi/lo pair (once).
// ---------------------------------------------------------------------------
__global__ void k_convert_x(const float* __restrict__ X,
                            __bf16* __restrict__ Xh, __bf16* __restrict__ Xl) {
  size_t i = (size_t)blockIdx.x * 256 + threadIdx.x;  // exactly N*D threads
  float x = X[i];
  __bf16 h = (__bf16)x;
  Xh[i] = h;
  Xl[i] = (__bf16)(x - (float)h);
}

// ---------------------------------------------------------------------------
// 2) Qt[m][d] = Q[d][m], plus bf16 hi/lo split.
// ---------------------------------------------------------------------------
__global__ void k_init_qt(const float* __restrict__ Q, float* __restrict__ Qtf,
                          __bf16* __restrict__ Qth, __bf16* __restrict__ Qtl) {
  int i = blockIdx.x * 256 + threadIdx.x;  // M*D = 16384 threads
  int m = i >> 8, d = i & 255;
  float q = Q[d * M + m];
  Qtf[i] = q;
  __bf16 h = (__bf16)q;
  Qth[i] = h;
  Qtl[i] = (__bf16)(q - (float)h);
}

// ---------------------------------------------------------------------------
// 3) GEMM1: S^T[m][n] = (X Q)[n][m], bf16x3 WMMA.
//    Grid: 512 blocks x 256 thr. Block stages full Qt hi+lo (64 KB) in LDS.
//    Wave w handles n-tile (blockIdx*8 + w): M=16 rows of n, all 64 m cols.
// ---------------------------------------------------------------------------
__global__ void __launch_bounds__(256)
k_gemm1(const __bf16* __restrict__ Xh, const __bf16* __restrict__ Xl,
        const __bf16* __restrict__ Qth, const __bf16* __restrict__ Qtl,
        float* __restrict__ St) {
  extern __shared__ __align__(16) char smem[];
  __bf16* qh = (__bf16*)smem;       // [64][256]
  __bf16* ql = qh + M * D;          // [64][256]

  int tid = threadIdx.x;
  {  // cooperative stage of Qt hi+lo (2 x 32 KB)
    const uint4* gh = (const uint4*)Qth;
    const uint4* gl = (const uint4*)Qtl;
    uint4* sh = (uint4*)qh;
    uint4* sl = (uint4*)ql;
#pragma unroll
    for (int i = tid; i < (M * D * 2 / 16); i += 256) { sh[i] = gh[i]; sl[i] = gl[i]; }
  }
  __syncthreads();

  int wave = tid >> 5, lane = tid & 31;
  int half = lane >> 4, r = lane & 15;            // A: M=r, K-half select
  size_t n0 = ((size_t)blockIdx.x * 8 + wave) * 16;

  const __bf16* arow_h = Xh + (n0 + r) * D;       // one X row per lane (M=r)
  const __bf16* arow_l = Xl + (n0 + r) * D;
  __builtin_prefetch(arow_h, 0, 3);
  __builtin_prefetch(arow_l, 0, 3);

  v8f acc[4] = {v8f{}, v8f{}, v8f{}, v8f{}};

#pragma unroll
  for (int k0 = 0; k0 < D; k0 += 32) {
    int kb = k0 + half * 16;  // contiguous A K order (matches B): lanes<16 K=0..15, lanes>=16 K=16..31
    v16bf a_hi = CAT16(*(const v8bf*)(arow_h + kb), *(const v8bf*)(arow_h + kb + 8));
    v16bf a_lo = CAT16(*(const v8bf*)(arow_l + kb), *(const v8bf*)(arow_l + kb + 8));
#pragma unroll
    for (int nt = 0; nt < 4; ++nt) {
      // B layout: N = lane&15, K = (lane<16 ? 0..15 : 16..31) + k0, contiguous in Qt row
      const __bf16* bhp = qh + (nt * 16 + r) * D + k0 + half * 16;
      const __bf16* blp = ql + (nt * 16 + r) * D + k0 + half * 16;
      v16bf b_hi = CAT16(((const v8bf*)bhp)[0], ((const v8bf*)bhp)[1]);
      v16bf b_lo = CAT16(((const v8bf*)blp)[0], ((const v8bf*)blp)[1]);
      acc[nt] = WMMA_BF16(a_hi, b_hi, acc[nt]);
      acc[nt] = WMMA_BF16(a_hi, b_lo, acc[nt]);
      acc[nt] = WMMA_BF16(a_lo, b_hi, acc[nt]);
    }
  }

  // C layout: lane -> N=r (m col), VGPR v -> M=v+8*half (n row). Per lane the 8
  // results land at 8 consecutive n in S^T row (nt*16+r) -> two b128 stores.
#pragma unroll
  for (int nt = 0; nt < 4; ++nt) {
    v8f a8 = acc[nt];
    float* p = St + (size_t)(nt * 16 + r) * N + n0 + half * 8;
    *(float4*)(p)     = ((float4*)&a8)[0];
    *(float4*)(p + 4) = ((float4*)&a8)[1];
  }
}

// ---------------------------------------------------------------------------
// 4) Sparsemax per column. One 1024-thread WG per m; whole 256 KB column in LDS,
//    DMA'd in by the Tensor Data Mover (falls back to VALU loads if the TDM
//    builtin is unavailable). tau iteration:
//    tau <- (sum_{z>tau} z - 1)/count_{z>tau}, from tau0 = zmax - 1
//    (monotone convergence to the exact sort-based threshold).
// ---------------------------------------------------------------------------
#if defined(__gfx1250__) && __has_builtin(__builtin_amdgcn_tensor_load_to_lds) && \
    __has_builtin(__builtin_amdgcn_s_wait_tensorcnt)
#define USE_TDM 1
#else
#define USE_TDM 0
#endif

__global__ void __launch_bounds__(1024)
k_sparsemax(const float* __restrict__ St, __bf16* __restrict__ Ph, __bf16* __restrict__ Pl) {
  extern __shared__ __align__(16) char smem[];
  float* col  = (float*)smem;     // [65536]
  float* redS = col + N;          // [1024]
  float* redC = redS + 1024;      // [1024]

  int tid = threadIdx.x;
  int m = blockIdx.x;
  const float* src = St + (size_t)m * N;

#if USE_TDM
  // --- TDM: DMA the whole 256 KB column (viewed as a 16 x 4096 fp32 tile,
  //     since tile dims are 16-bit fields) straight into LDS. One TDM issue
  //     from wave 0; TENSORcnt-tracked; no VALU/VGPR involvement.
  if (tid < 32) {
    unsigned long long ga = (unsigned long long)(const void*)src;
    unsigned int lds_byte = (unsigned int)(unsigned long long)(void*)col;  // addr[31:0] = LDS offset
    // D# group 0: count=1 | lds_addr | global_addr[56:0] | type=2
    u32x4 g0;
    g0[0] = 1u;                                   // count=1, user mode, no gather
    g0[1] = lds_byte;                             // lds_addr (bytes)
    g0[2] = (unsigned int)(ga & 0xffffffffu);     // global_addr[31:0]
    g0[3] = (unsigned int)((ga >> 32) & 0x01ffffffu) | (2u << 30);  // addr[56:32] | type=2
    // D# group 1: data_size=4B; tensor 4096 x 16; tile 4096 x 16; stride0=4096
    i32x8 g1;
    g1[0] = (int)(2u << 16);                      // wg_mask=0, data_size=2 (4 bytes)
    g1[1] = (int)(4096u << 16);                   // tensor_dim0[15:0]=4096 (bits 63:48)
    g1[2] = (int)(16u << 16);                     // dim0[31:16]=0, tensor_dim1[15:0]=16
    g1[3] = (int)(4096u << 16);                   // dim1[31:16]=0, tile_dim0=4096
    g1[4] = 16;                                   // tile_dim1=16, tile_dim2=0
    g1[5] = 4096;                                 // tensor_dim0_stride[31:0]
    g1[6] = 0;                                    // stride0[47:32]=0, stride1[15:0]=0
    g1[7] = 0;
    i32x4 gz = {0, 0, 0, 0};                      // groups 2/3 unused (<=2D tensor)
#if __clang_major__ >= 23
    __builtin_amdgcn_tensor_load_to_lds(g0, g1, gz, gz, (i32x8){0,0,0,0,0,0,0,0}, 0);
#else
    __builtin_amdgcn_tensor_load_to_lds(g0, g1, gz, gz, 0);
#endif
    __builtin_amdgcn_s_wait_tensorcnt(0);
  }
  __syncthreads();

  float lmax = -3.4e38f;
#pragma unroll 4
  for (int j = 0; j < N / 1024; ++j) lmax = fmaxf(lmax, col[tid + j * 1024]);
#else
  float lmax = -3.4e38f;
#pragma unroll 4
  for (int j = 0; j < N / 1024; ++j) {
    int i = tid + j * 1024;
    float v = src[i];
    col[i] = v;
    lmax = fmaxf(lmax, v);
  }
#endif

  redS[tid] = lmax;
  __syncthreads();
  for (int s = 512; s > 0; s >>= 1) {
    if (tid < s) redS[tid] = fmaxf(redS[tid], redS[tid + s]);
    __syncthreads();
  }
  float tau = redS[0] - 1.0f;
  __syncthreads();

  for (int it = 0; it < 24; ++it) {
    float s = 0.f, c = 0.f;
#pragma unroll 4
    for (int j = 0; j < N / 1024; ++j) {
      float v = col[tid + j * 1024];
      if (v > tau) { s += v; c += 1.f; }
    }
    redS[tid] = s;
    redC[tid] = c;
    __syncthreads();
    for (int st = 512; st > 0; st >>= 1) {
      if (tid < st) { redS[tid] += redS[tid + st]; redC[tid] += redC[tid + st]; }
      __syncthreads();
    }
    tau = (redS[0] - 1.0f) / redC[0];
    __syncthreads();
  }

#pragma unroll 4
  for (int j = 0; j < N / 1024; ++j) {
    int i = tid + j * 1024;
    float p = col[i] - tau;
    p = p > 0.f ? p : 0.f;
    __bf16 h = (__bf16)p;
    Ph[(size_t)m * N + i] = h;
    Pl[(size_t)m * N + i] = (__bf16)(p - (float)h);
  }
}

// ---------------------------------------------------------------------------
// 5) GEMM2 split-K: partial[chunk][m][d] = sum_{n in chunk} p^T[m][n] X[n][d].
//    64 chunks of 1024 n-rows. X chunk transposed through LDS for the B operand.
//    Wave w: m-tile = w&3, d-tiles = (w>>2)*8 .. +7 (8 accumulators).
// ---------------------------------------------------------------------------
__global__ void __launch_bounds__(256)
k_gemm2(const __bf16* __restrict__ Xh, const __bf16* __restrict__ Xl,
        const __bf16* __restrict__ Ph, const __bf16* __restrict__ Pl,
        float* __restrict__ partials) {
  extern __shared__ __align__(16) char smem[];
  __bf16* xth = (__bf16*)smem;     // [256 d][32 k]
  __bf16* xtl = xth + D * 32;      // [256 d][32 k]

  int tid = threadIdx.x;
  int wave = tid >> 5, lane = tid & 31;
  int half = lane >> 4, r = lane & 15;
  int chunk = blockIdx.x;
  size_t nbase = (size_t)chunk * 1024;
  int mt = wave & 3, dh = wave >> 2;

  const __bf16* aprow_h = Ph + (size_t)(mt * 16 + r) * N + nbase;  // A row: M = m
  const __bf16* aprow_l = Pl + (size_t)(mt * 16 + r) * N + nbase;

  v8f acc[8] = {v8f{}, v8f{}, v8f{}, v8f{}, v8f{}, v8f{}, v8f{}, v8f{}};

  for (int ks = 0; ks < 32; ++ks) {
    size_t n0 = nbase + ks * 32;
    __syncthreads();  // previous tile fully consumed before overwrite
    {  // transpose-stage 32 X rows (hi+lo) into LDS: xt[d][k]
      int nr = tid >> 3;             // 0..31  (k index within chunk step)
      int dseg = (tid & 7) * 32;     // 8 threads cover a 256-wide row
      const __bf16* grh = Xh + (n0 + nr) * D + dseg;
      const __bf16* grl = Xl + (n0 + nr) * D + dseg;
#pragma unroll
      for (int e = 0; e < 32; ++e) {
        xth[(dseg + e) * 32 + nr] = grh[e];
        xtl[(dseg + e) * 32 + nr] = grl[e];
      }
    }
    __syncthreads();

    int kb = ks * 32 + half * 16;
    v16bf a_hi = CAT16(*(const v8bf*)(aprow_h + kb), *(const v8bf*)(aprow_h + kb + 8));
    v16bf a_lo = CAT16(*(const v8bf*)(aprow_l + kb), *(const v8bf*)(aprow_l + kb + 8));
#pragma unroll
    for (int t = 0; t < 8; ++t) {
      int d = (dh * 8 + t) * 16 + r;  // B: N = d
      const __bf16* bhp = xth + d * 32 + half * 16;
      const __bf16* blp = xtl + d * 32 + half * 16;
      v16bf b_hi = CAT16(((const v8bf*)bhp)[0], ((const v8bf*)bhp)[1]);
      v16bf b_lo = CAT16(((const v8bf*)blp)[0], ((const v8bf*)blp)[1]);
      acc[t] = WMMA_BF16(a_hi, b_hi, acc[t]);
      acc[t] = WMMA_BF16(a_hi, b_lo, acc[t]);
      acc[t] = WMMA_BF16(a_lo, b_hi, acc[t]);
    }
  }

#pragma unroll
  for (int t = 0; t < 8; ++t) {
    int d = (dh * 8 + t) * 16 + r;
#pragma unroll
    for (int v = 0; v < 8; ++v) {
      int mrow = mt * 16 + v + half * 8;
      partials[((size_t)chunk * M + mrow) * D + d] = acc[t][v];
    }
  }
}

// ---------------------------------------------------------------------------
// 6) Deterministic split-K reduce: Qt = sum over 64 chunks; refresh hi/lo.
// ---------------------------------------------------------------------------
__global__ void k_finalize(const float* __restrict__ partials, float* __restrict__ Qtf,
                           __bf16* __restrict__ Qth, __bf16* __restrict__ Qtl) {
  int i = blockIdx.x * 256 + threadIdx.x;  // M*D threads
  float s = 0.f;
#pragma unroll 8
  for (int c = 0; c < 64; ++c) s += partials[(size_t)c * (M * D) + i];
  Qtf[i] = s;
  __bf16 h = (__bf16)s;
  Qth[i] = h;
  Qtl[i] = (__bf16)(s - (float)h);
}

// ---------------------------------------------------------------------------
// 7) out[d][m] = Qt[m][d]
// ---------------------------------------------------------------------------
__global__ void k_writeout(const float* __restrict__ Qtf, float* __restrict__ out) {
  int i = blockIdx.x * 256 + threadIdx.x;  // M*D threads
  int m = i >> 8, d = i & 255;
  out[d * M + m] = Qtf[i];
}

// ---------------------------------------------------------------------------
extern "C" void kernel_launch(void* const* d_in, const int* in_sizes, int n_in,
                              void* d_out, int out_size, void* d_ws, size_t ws_size,
                              hipStream_t stream) {
  const float* Q = (const float*)d_in[0];  // [256,64]
  const float* X = (const float*)d_in[1];  // [65536,256]
  float* out = (float*)d_out;

  char* ws = (char*)d_ws;
  __bf16* Xh   = (__bf16*)(ws + OFF_XH);
  __bf16* Xl   = (__bf16*)(ws + OFF_XL);
  float*  St   = (float*)(ws + OFF_ST);
  __bf16* Ph   = (__bf16*)(ws + OFF_PH);
  __bf16* Pl   = (__bf16*)(ws + OFF_PL);
  float*  Part = (float*)(ws + OFF_PART);
  float*  Qtf  = (float*)(ws + OFF_QTF);
  __bf16* Qth  = (__bf16*)(ws + OFF_QTH);
  __bf16* Qtl  = (__bf16*)(ws + OFF_QTL);

  k_convert_x<<<(N * D) / 256, 256, 0, stream>>>(X, Xh, Xl);
  k_init_qt<<<(M * D) / 256, 256, 0, stream>>>(Q, Qtf, Qth, Qtl);

  const size_t lds_g1 = (size_t)M * D * 2 * 2;          // 64 KB  (Qt hi+lo)
  const size_t lds_sm = (size_t)N * 4 + 2 * 1024 * 4;   // 264 KB (column + reduce)
  const size_t lds_g2 = (size_t)D * 32 * 2 * 2;         // 32 KB  (Xt hi+lo)

  for (int it = 0; it < ITERS; ++it) {
    k_gemm1<<<(N / 16) / 8, 256, lds_g1, stream>>>(Xh, Xl, Qth, Qtl, St);
    k_sparsemax<<<M, 1024, lds_sm, stream>>>(St, Ph, Pl);
    k_gemm2<<<64, 256, lds_g2, stream>>>(Xh, Xl, Ph, Pl, Part);
    k_finalize<<<(M * D) / 256, 256, 0, stream>>>(Part, Qtf, Qth, Qtl);
  }

  k_writeout<<<(M * D) / 256, 256, 0, stream>>>(Qtf, out);
}
